// TorchPatchNN_45689862095502
// MI455X (gfx1250) — compile-verified, matrix operand order checked by reading, not run on hardware
//
#include <hip/hip_runtime.h>
#include <hip/hip_bf16.h>

// ---------------------------------------------------------------------------
// TorchPatchNN on MI455X (gfx1250): fused WMMA GEMM + argmin, split-N version
//   dist(m,n) = |q_m|^2 + |k_n|^2 - 2 * <q_m, k_n>
// The 8100x8100 distance matrix (262 MB) is never materialized. Each wave owns
// 16 query rows and one of 4 contiguous key ranges; it sweeps its range in
// 16-wide tiles with v_wmma_f32_16x16x32_f16 keeping a running (min, argmin)
// in registers. f32 fidelity via hi/lo f16 split: 3 chained WMMAs per K-chunk.
// A second tiny kernel reduces the 4 partial argmins per query row.
// (507 x 4) waves ~= 2 waves per SIMD32 chip-wide for latency hiding, vs 507
// lone waves in the unsplit version.
// ---------------------------------------------------------------------------

typedef __attribute__((ext_vector_type(16))) _Float16 v16h;
typedef __attribute__((ext_vector_type(8)))  float    v8f;

#define NQ      8100    // query patches (90*90)
#define NK      8100    // key rows
#define NPAD    8112    // padded to multiple of 16 (507 tiles)
#define NTILES  (NPAD / 16)
#define NSPLIT  4       // key-range splits for occupancy
#define DDIM    147     // 3*7*7
#define DP      160     // padded K-dim: 5 chunks of 32
#define KCH     5       // DP/32
#define IMH     96
#define IMW     96
#define NCH     3
#define HO      90
#define WO      90
#define PH_     7
#define PW_     7

// ---------------------------------------------------------------- stage A ---
// Unfold query image into padded hi/lo f16 patch rows [NPAD][DP].
__global__ void k_unfold_split(const float* __restrict__ query,
                               _Float16* __restrict__ qh,
                               _Float16* __restrict__ ql) {
    const int p = blockIdx.x;      // patch id 0..NPAD-1
    const int d = threadIdx.x;     // 0..DP-1
    float v = 0.0f;
    if (p < NQ && d < DDIM) {
        const int c  = d / 49;
        const int r  = (d % 49) / 7;
        const int s  = d % 7;
        const int pi = p / WO;
        const int pj = p % WO;
        v = query[(c * IMH + pi + r) * IMW + (pj + s)];
    }
    const _Float16 hi = (_Float16)v;
    const _Float16 lo = (_Float16)(v - (float)hi);
    const size_t o = (size_t)p * DP + d;
    qh[o] = hi;
    ql[o] = lo;
}

// f32 query-patch norms straight from the image (full precision).
__global__ void k_qnorm(const float* __restrict__ query,
                        float* __restrict__ qn) {
    const int p = blockIdx.x * blockDim.x + threadIdx.x;
    if (p >= NPAD) return;
    float s = 0.0f;
    if (p < NQ) {
        const int pi = p / WO, pj = p % WO;
        for (int c = 0; c < NCH; ++c)
            for (int r = 0; r < PH_; ++r)
                for (int q2 = 0; q2 < PW_; ++q2) {
                    const float v = query[(c * IMH + pi + r) * IMW + (pj + q2)];
                    s += v * v;
                }
    }
    qn[p] = s;
}

// Split key matrix into padded hi/lo f16 [NPAD][DP].
__global__ void k_key_split(const float* __restrict__ key,
                            _Float16* __restrict__ kh,
                            _Float16* __restrict__ kl) {
    const int r = blockIdx.x;
    const int d = threadIdx.x;
    float v = 0.0f;
    if (r < NK && d < DDIM) v = key[(size_t)r * DDIM + d];
    const _Float16 hi = (_Float16)v;
    const _Float16 lo = (_Float16)(v - (float)hi);
    const size_t o = (size_t)r * DP + d;
    kh[o] = hi;
    kl[o] = lo;
}

// f32 key norms; padded rows get +huge so they can never win the argmin.
__global__ void k_knorm(const float* __restrict__ key,
                        float* __restrict__ kn) {
    const int r = blockIdx.x * blockDim.x + threadIdx.x;
    if (r >= NPAD) return;
    float s = 1.0e30f;
    if (r < NK) {
        s = 0.0f;
        for (int d = 0; d < DDIM; ++d) {
            const float v = key[(size_t)r * DDIM + d];
            s += v * v;
        }
    }
    kn[r] = s;
}

// ---------------------------------------------------------------- stage B ---
// One wave (32 lanes) per (16-query M-tile, key-range split). A fragments
// (hi+lo, 5 chunks) are preloaded and reused across all key tiles in the
// split. Per tile: 15 chained WMMAs (3-product hi/lo split x 5 chunks), then
// per-lane running min/argmin.
// C layout: lane = column n (lane&15), VGPR r = row m (r + 8*(lane>=16)).
__global__ __launch_bounds__(32)
void k_wmma_argmin(const _Float16* __restrict__ qh,
                   const _Float16* __restrict__ ql,
                   const _Float16* __restrict__ kh,
                   const _Float16* __restrict__ kl,
                   const float* __restrict__ qn,
                   const float* __restrict__ kn,
                   float* __restrict__ pmv,     // [NSPLIT][NPAD] partial min
                   int* __restrict__ pmi) {     // [NSPLIT][NPAD] partial idx
    const int lane  = threadIdx.x;
    const int m0    = blockIdx.x * 16;
    const int split = blockIdx.y;
    const int rsel  = lane & 15;           // A row / B column / C column slot
    const int hi8   = (lane >> 4) & 1;     // lane half
    const int kb    = hi8 ? 8 : 0;         // K sub-run select per ISA layout

    // --- preload A fragments: 16-bit A 16x32 layout, two b128 runs per chunk
    v16h aH[KCH], aL[KCH];
    {
        const _Float16* rowH = qh + (size_t)(m0 + rsel) * DP;
        const _Float16* rowL = ql + (size_t)(m0 + rsel) * DP;
#pragma unroll
        for (int c = 0; c < KCH; ++c) {
            const int kbase = c * 32 + kb;
            ((uint4*)&aH[c])[0] = *(const uint4*)(rowH + kbase);
            ((uint4*)&aH[c])[1] = *(const uint4*)(rowH + kbase + 16);
            ((uint4*)&aL[c])[0] = *(const uint4*)(rowL + kbase);
            ((uint4*)&aL[c])[1] = *(const uint4*)(rowL + kbase + 16);
        }
    }

    // query norms for the 8 C rows this lane owns
    float qrm[8];
#pragma unroll
    for (int r = 0; r < 8; ++r) qrm[r] = qn[m0 + hi8 * 8 + r];

    float mv[8];
    int   mi[8];
#pragma unroll
    for (int r = 0; r < 8; ++r) { mv[r] = 3.4e38f; mi[r] = 0; }

    const int nt0 = (NTILES * split) / NSPLIT;
    const int nt1 = (NTILES * (split + 1)) / NSPLIT;
    for (int nt = nt0; nt < nt1; ++nt) {
        const int n0   = nt * 16;
        const int ncol = n0 + rsel;
        const _Float16* krowH = kh + (size_t)ncol * DP;
        const _Float16* krowL = kl + (size_t)ncol * DP;
        // prefetch the next tile's key rows into cache (global_prefetch_b8)
        if (nt + 1 < nt1) {
            __builtin_prefetch(krowH + 16 * DP, 0, 1);
            __builtin_prefetch(krowL + 16 * DP, 0, 1);
        }

        v8f acc = {};
#pragma unroll
        for (int c = 0; c < KCH; ++c) {
            const int kbase = c * 32 + kb;
            v16h bH, bL;
            ((uint4*)&bH)[0] = *(const uint4*)(krowH + kbase);
            ((uint4*)&bH)[1] = *(const uint4*)(krowH + kbase + 16);
            ((uint4*)&bL)[0] = *(const uint4*)(krowL + kbase);
            ((uint4*)&bL)[1] = *(const uint4*)(krowL + kbase + 16);
            // dot ~= qL*kH + qH*kL + qH*kH  (chained f32 accumulation)
            acc = __builtin_amdgcn_wmma_f32_16x16x32_f16(
                      false, aL[c], false, bH, (short)0, acc, false, false);
            acc = __builtin_amdgcn_wmma_f32_16x16x32_f16(
                      false, aH[c], false, bL, (short)0, acc, false, false);
            acc = __builtin_amdgcn_wmma_f32_16x16x32_f16(
                      false, aH[c], false, bH, (short)0, acc, false, false);
        }

        const float knv = kn[ncol];
#pragma unroll
        for (int r = 0; r < 8; ++r) {
            const float dist = qrm[r] + knv - 2.0f * acc[r];
            if (dist < mv[r]) { mv[r] = dist; mi[r] = ncol; }
        }
    }

    // cross-lane min reduction over the 16 column slots (stays in lane half:
    // xor of bits 0..3 never crosses bit 4). Tie-break on smaller index.
#pragma unroll
    for (int off = 1; off < 16; off <<= 1) {
#pragma unroll
        for (int r = 0; r < 8; ++r) {
            const float ov = __shfl_xor(mv[r], off, 32);
            const int   oi = __shfl_xor(mi[r], off, 32);
            if (ov < mv[r] || (ov == mv[r] && oi < mi[r])) {
                mv[r] = ov; mi[r] = oi;
            }
        }
    }

    if (rsel == 0) {
        const size_t base = (size_t)split * NPAD + m0 + hi8 * 8;
#pragma unroll
        for (int r = 0; r < 8; ++r) {
            pmv[base + r] = mv[r];
            pmi[base + r] = mi[r];
        }
    }
}

// Combine the NSPLIT partial (min, argmin) per query row. Splits are in
// ascending key order, so lexicographic (val, idx) min preserves the
// first-minimum tie-break of jnp.argmin.
__global__ void k_reduce_splits(const float* __restrict__ pmv,
                                const int* __restrict__ pmi,
                                int* __restrict__ idx) {
    const int m = blockIdx.x * blockDim.x + threadIdx.x;
    if (m >= NQ) return;
    float bv = pmv[m];
    int   bi = pmi[m];
#pragma unroll
    for (int s = 1; s < NSPLIT; ++s) {
        const float v = pmv[(size_t)s * NPAD + m];
        const int   i = pmi[(size_t)s * NPAD + m];
        if (v < bv || (v == bv && i < bi)) { bv = v; bi = i; }
    }
    idx[m] = bi;
}

// ---------------------------------------------------------------- stage C ---
// Gather matched value patches and fold back with overlap-add mean.
__global__ void k_fold_mean(const float* __restrict__ value,
                            const int* __restrict__ idx,
                            float* __restrict__ out) {
    const int t = blockIdx.x * blockDim.x + threadIdx.x;
    if (t >= NCH * IMH * IMW) return;
    const int x = t % IMW;
    const int y = (t / IMW) % IMH;
    const int c = t / (IMW * IMH);

    int di_lo = y - (HO - 1); if (di_lo < 0) di_lo = 0;
    int di_hi = y;            if (di_hi > PH_ - 1) di_hi = PH_ - 1;
    int dj_lo = x - (WO - 1); if (dj_lo < 0) dj_lo = 0;
    int dj_hi = x;            if (dj_hi > PW_ - 1) dj_hi = PW_ - 1;

    float s = 0.0f;
    int cnt = 0;
    for (int di = di_lo; di <= di_hi; ++di) {
        const int pi = y - di;
        for (int dj = dj_lo; dj <= dj_hi; ++dj) {
            const int pj = x - dj;
            const int p  = pi * WO + pj;
            s += value[(size_t)idx[p] * DDIM + (c * 49 + di * 7 + dj)];
            ++cnt;
        }
    }
    out[t] = s / (float)cnt;
}

// ---------------------------------------------------------------------------
extern "C" void kernel_launch(void* const* d_in, const int* in_sizes, int n_in,
                              void* d_out, int out_size, void* d_ws, size_t ws_size,
                              hipStream_t stream) {
    const float* query = (const float*)d_in[0];   // [1,3,96,96]
    const float* key   = (const float*)d_in[1];   // [8100,147]
    const float* value = (const float*)d_in[2];   // [8100,147]
    float* out = (float*)d_out;                   // [1,3,96,96]

    // workspace layout (each half-array is 2,595,840 B, 256B-aligned)
    char* ws = (char*)d_ws;
    const size_t HBYTES = (size_t)NPAD * DP * sizeof(_Float16);
    _Float16* qh = (_Float16*)(ws);
    _Float16* ql = (_Float16*)(ws + 1 * HBYTES);
    _Float16* kh = (_Float16*)(ws + 2 * HBYTES);
    _Float16* kl = (_Float16*)(ws + 3 * HBYTES);
    float*    qn  = (float*)(ws + 4 * HBYTES);
    float*    kn  = qn + NPAD;
    float*    pmv = kn + NPAD;                 // [NSPLIT][NPAD]
    int*      pmi = (int*)(pmv + NSPLIT * NPAD);
    int*      idx = pmi + NSPLIT * NPAD;

    k_unfold_split<<<NPAD, DP, 0, stream>>>(query, qh, ql);
    k_qnorm<<<(NPAD + 255) / 256, 256, 0, stream>>>(query, qn);
    k_key_split<<<NPAD, DP, 0, stream>>>(key, kh, kl);
    k_knorm<<<(NPAD + 255) / 256, 256, 0, stream>>>(key, kn);

    dim3 grid(NTILES, NSPLIT);
    k_wmma_argmin<<<grid, 32, 0, stream>>>(qh, ql, kh, kl, qn, kn, pmv, pmi);
    k_reduce_splits<<<(NQ + 255) / 256, 256, 0, stream>>>(pmv, pmi, idx);

    k_fold_mean<<<(NCH * IMH * IMW + 255) / 256, 256, 0, stream>>>(value, idx, out);
}
